// PrimalGCNNet_69724499083523
// MI455X (gfx1250) — matrix operand order repaired and checked by reading, not compiled, not run on hardware
//
#include <hip/hip_runtime.h>
#include <hip/hip_bf16.h>

#define NEQ   4096
#define NGEN  2048
#define XDIMC (NEQ + NGEN)   // 6144
#define BATCH 512
#define H1C   2048
#define H2C   2048
#define YDIMC 4096
#define NEDGE 32768

typedef __attribute__((ext_vector_type(16))) __bf16 v16bf;
typedef __attribute__((ext_vector_type(8)))  __bf16 v8bf;
typedef __attribute__((ext_vector_type(8)))  float  v8f;

__device__ __forceinline__ __bf16 f2bf(float f) {
  unsigned u = __builtin_bit_cast(unsigned, f);
  unsigned r = u + 0x7FFFu + ((u >> 16) & 1u);   // round-to-nearest-even
  unsigned short h = (unsigned short)(r >> 16);
  return __builtin_bit_cast(__bf16, h);
}

// ---------------- GCN front-end (tiny) ----------------

__global__ void k_zero(unsigned* cnt, float* agg) {
  int i = blockIdx.x * 256 + threadIdx.x;
  if (i < NEQ) { cnt[i] = 0u; agg[i] = 0.0f; }
}

__global__ void k_count(const int* __restrict__ ei, unsigned* cnt) {
  int e = blockIdx.x * 256 + threadIdx.x;
  if (e < NEDGE) atomicAdd(&cnt[ei[NEDGE + e]], 1u);
}

__global__ void k_dinv(const unsigned* __restrict__ cnt, float* dinv) {
  int i = blockIdx.x * 256 + threadIdx.x;
  if (i < NEQ) dinv[i] = rsqrtf(1.0f + 512.0f * (float)cnt[i]);
}

__global__ void k_edge(const int* __restrict__ ei, const float* __restrict__ x,
                       const float* __restrict__ dinv, float* agg) {
  int e = blockIdx.x * 256 + threadIdx.x;
  if (e >= NEDGE) return;
  int s = ei[e];             // src node id < NEQ : node i = n*512 + b
  int d = ei[NEDGE + e];     // dst node id < NEQ
  float xv = x[(s & 511) * XDIMC + (s >> 9)];   // xf[s] = x[b_s, n_s]
  atomicAdd(&agg[d], dinv[s] * xv);
}

// build z[b][n] in bf16: relu(GCN) for n<NEQ, raw x_g for n>=NEQ
__global__ void k_buildz(const float* __restrict__ x, const float* __restrict__ dinv,
                         const float* __restrict__ agg, const float* __restrict__ gw,
                         const float* __restrict__ gb, __bf16* __restrict__ z) {
  int n = blockIdx.x * 256 + threadIdx.x;
  int b = blockIdx.y;
  float xv = x[b * XDIMC + n];
  float v;
  if (n < NEQ) {
    float w = gw[0], bias = gb[0];
    float h;
    if (n < 8) {                           // only node ids < NEQ get edges
      int i = n * 512 + b;
      float di = dinv[i];
      h = w * (di * di * xv + 512.0f * di * agg[i]) + bias;
    } else {                               // deg==1 self-loop only
      h = w * xv + bias;
    }
    v = fmaxf(h, 0.0f);
  } else {
    v = xv;                                // x_g passthrough, no relu
  }
  z[b * XDIMC + n] = f2bf(v);
}

// transpose + convert: in f32 [K][N]  ->  out bf16 [N][K]
// 32x32 tiles through LDS; both global accesses coalesced.
__global__ __launch_bounds__(256) void k_cvt_t(const float* __restrict__ in,
                                               __bf16* __restrict__ out,
                                               int K, int N) {
  __shared__ float tile[32][33];
  const int nt = blockIdx.x * 32;
  const int kt = blockIdx.y * 32;
  const int tx = threadIdx.x & 31;
  const int ty = threadIdx.x >> 5;   // 8 rows per pass
#pragma unroll
  for (int i = 0; i < 4; ++i) {
    int k = ty + i * 8;
    tile[k][tx] = in[(long)(kt + k) * N + nt + tx];
  }
  __syncthreads();
#pragma unroll
  for (int i = 0; i < 4; ++i) {
    int n = ty + i * 8;
    out[(long)(nt + n) * K + kt + tx] = f2bf(tile[tx][n]);
  }
}

// ---------------- bf16 WMMA GEMM: Out = act(A[M,K] @ Bt[N,K]^T + bias) ------------
// 128 thr = 4 waves (2x2). Block tile 128x128, BK=64, TRIPLE-buffered LDS.
// Both tiles stream via global_load_async_to_lds_b128 (ASYNCcnt, in-order
// completion). Prefetch distance 2 tiles; partial wait s_wait_asynccnt 16
// keeps the newest tile's 16 ops/thread in flight across the barrier.
// Each wave: 64x64 sub-tile; 2 K-steps x (4x4) v_wmma_f32_16x16x32_bf16.

template<bool RELU, bool OUT_BF16>
__global__ __launch_bounds__(128) void k_gemm(const __bf16* __restrict__ A,
                                              const __bf16* __restrict__ Bt,
                                              const float* __restrict__ bias,
                                              void* __restrict__ Out,
                                              int M, int N, int K) {
  constexpr int BK = 64, PAD = 8, LDW = BK + PAD;   // row stride 144B (16B mult)
  __shared__ __align__(16) __bf16 As[3][128][LDW];  // A tile,  row = m, col = k
  __shared__ __align__(16) __bf16 Bs[3][128][LDW];  // Bt tile, row = n, col = k

  const int tid  = threadIdx.x;
  const int lane = tid & 31;
  const int wave = tid >> 5;
  const int wrow = wave >> 1;   // 0..1
  const int wcol = wave & 1;    // 0..1
  const int m0 = blockIdx.y * 128;
  const int n0 = blockIdx.x * 128;

  v8f acc[4][4];
#pragma unroll
  for (int mi = 0; mi < 4; ++mi)
#pragma unroll
    for (int ni = 0; ni < 4; ++ni)
      acc[mi][ni] = (v8f){0.f, 0.f, 0.f, 0.f, 0.f, 0.f, 0.f, 0.f};

  // async copy of a 128x64 bf16 tile (row-major, leading dim `ld`) into LDS
  // (8 x global_load_async_to_lds_b128 per thread per matrix)
  auto asyncTile = [&](const __bf16* src, long ld, int row0, int kt,
                       __bf16 (*dst)[LDW]) {
#pragma unroll
    for (int it = 0; it < 8; ++it) {
      int l8 = (it * 128 + tid) * 8;
      int r = l8 >> 6, c = l8 & 63;
      unsigned long long gp =
          (unsigned long long)(const void*)&src[(long)(row0 + r) * ld + kt + c];
      unsigned lp = (unsigned)(unsigned long long)(const void*)&dst[r][c];
      asm volatile("global_load_async_to_lds_b128 %0, %1, off"
                   :: "v"(lp), "v"(gp) : "memory");
    }
  };

  // prologue: stage tiles 0 and 1 (tile 0 guaranteed complete by partial wait)
  asyncTile(A,  K, m0, 0, As[0]);
  asyncTile(Bt, K, n0, 0, Bs[0]);
  if (BK < K) {
    asyncTile(A,  K, m0, BK, As[1]);
    asyncTile(Bt, K, n0, BK, Bs[1]);
    asm volatile("s_wait_asynccnt 0x10" ::: "memory");   // tile 0 done (in-order)
  } else {
    asm volatile("s_wait_asynccnt 0x0" ::: "memory");
  }
  __syncthreads();

  int buf = 0;
  for (int kt = 0; kt < K; kt += BK) {
    const bool pf2 = (kt + 2 * BK) < K;
    if (pf2) {                       // prefetch two tiles ahead
      int pb = buf + 2; if (pb >= 3) pb -= 3;
      asyncTile(A,  K, m0, kt + 2 * BK, As[pb]);
      asyncTile(Bt, K, n0, kt + 2 * BK, Bs[pb]);
    }

#pragma unroll
    for (int kk = 0; kk < BK; kk += 32) {
      // A fragment (16-bit A 16x32): lane L row M=L&15; half=L>>4;
      //   elems j<8 -> K = half*8 + j ; elems j>=8 -> K = 16 + half*8 + (j-8)
      v16bf af[4], bfr[4];
      const int c = (lane >> 4) * 8;
#pragma unroll
      for (int mi = 0; mi < 4; ++mi) {
        int rm = wrow * 64 + mi * 16 + (lane & 15);
        v8bf lo = *(const v8bf*)&As[buf][rm][kk + c];
        v8bf hi = *(const v8bf*)&As[buf][rm][kk + 16 + c];
        af[mi] = __builtin_shufflevector(lo, hi, 0, 1, 2, 3, 4, 5, 6, 7,
                                         8, 9, 10, 11, 12, 13, 14, 15);
      }
      // B fragment (32x16): lane L col N=L&15; j<8 -> K = (L>>4)*8 + j ; j>=8 -> K = 16 + (L>>4)*8 + (j-8)
      const int ks = (lane >> 4) * 8;
#pragma unroll
      for (int ni = 0; ni < 4; ++ni) {
        int cn = wcol * 64 + ni * 16 + (lane & 15);
        v8bf lo = *(const v8bf*)&Bs[buf][cn][kk + ks];
        v8bf hi = *(const v8bf*)&Bs[buf][cn][kk + 16 + ks];
        bfr[ni] = __builtin_shufflevector(lo, hi, 0, 1, 2, 3, 4, 5, 6, 7,
                                          8, 9, 10, 11, 12, 13, 14, 15);
      }
#pragma unroll
      for (int mi = 0; mi < 4; ++mi)
#pragma unroll
        for (int ni = 0; ni < 4; ++ni)
          acc[mi][ni] = __builtin_amdgcn_wmma_f32_16x16x32_bf16(
              false, af[mi], false, bfr[ni], (short)0, acc[mi][ni], false, false);
    }

    // ensure tile kt+BK resident before next iteration; keep tile kt+2BK
    // (16 ops/thread) in flight. ASYNCcnt completes in order.
    if (pf2)
      asm volatile("s_wait_asynccnt 0x10" ::: "memory");
    else
      asm volatile("s_wait_asynccnt 0x0" ::: "memory");
    __syncthreads();
    buf = (buf + 1 < 3) ? buf + 1 : 0;
  }

  // D layout: VGPR r -> row m = r + (lane>>4)*8 ; col n = lane&15
#pragma unroll
  for (int ni = 0; ni < 4; ++ni) {
    int n = n0 + wcol * 64 + ni * 16 + (lane & 15);
    float bv = bias[n];
#pragma unroll
    for (int mi = 0; mi < 4; ++mi) {
      int mbase = m0 + wrow * 64 + mi * 16 + ((lane >> 4) << 3);
#pragma unroll
      for (int r = 0; r < 8; ++r) {
        float v = acc[mi][ni][r] + bv;
        if (RELU) v = fmaxf(v, 0.0f);
        long off = (long)(mbase + r) * N + n;
        if (OUT_BF16) ((__bf16*)Out)[off] = f2bf(v);
        else          ((float*)Out)[off]  = v;
      }
    }
  }
}

// ---------------- launch ----------------

extern "C" void kernel_launch(void* const* d_in, const int* in_sizes, int n_in,
                              void* d_out, int out_size, void* d_ws, size_t ws_size,
                              hipStream_t stream) {
  const float* x  = (const float*)d_in[0];
  const int*   ei = (const int*)d_in[1];
  const float* gw = (const float*)d_in[2];
  const float* gb = (const float*)d_in[3];
  const float* W1 = (const float*)d_in[4];
  const float* b1 = (const float*)d_in[5];
  const float* W2 = (const float*)d_in[6];
  const float* b2 = (const float*)d_in[7];
  const float* W3 = (const float*)d_in[8];
  const float* b3 = (const float*)d_in[9];

  char* ws = (char*)d_ws;
  unsigned* cnt  = (unsigned*)(ws);
  float*    dinv = (float*)(ws + 16384);
  float*    agg  = (float*)(ws + 32768);
  __bf16*   z    = (__bf16*)(ws + 49152);      // 512x6144 bf16   [M][K]
  __bf16*   z1   = (__bf16*)(ws + 6340608);    // 512x2048 bf16
  __bf16*   z2   = (__bf16*)(ws + 8437760);    // 512x2048 bf16
  __bf16*   W1t  = (__bf16*)(ws + 10534912);   // 2048x6144 bf16  [N][K]
  __bf16*   W2t  = (__bf16*)(ws + 35700736);   // 2048x2048 bf16  [N][K]
  __bf16*   W3t  = (__bf16*)(ws + 44089344);   // 4096x2048 bf16  [N][K]

  // GCN front-end
  k_zero <<<16, 256, 0, stream>>>(cnt, agg);
  k_count<<<NEDGE / 256, 256, 0, stream>>>(ei, cnt);
  k_dinv <<<16, 256, 0, stream>>>(cnt, dinv);
  k_edge <<<NEDGE / 256, 256, 0, stream>>>(ei, x, dinv, agg);
  k_buildz<<<dim3(XDIMC / 256, BATCH), 256, 0, stream>>>(x, dinv, agg, gw, gb, z);

  // weight transpose+conversion f32 [K][N] -> bf16 [N][K]
  k_cvt_t<<<dim3(H1C / 32,  XDIMC / 32), 256, 0, stream>>>(W1, W1t, XDIMC, H1C);
  k_cvt_t<<<dim3(H2C / 32,  H1C / 32),   256, 0, stream>>>(W2, W2t, H1C,  H2C);
  k_cvt_t<<<dim3(YDIMC / 32, H2C / 32),  256, 0, stream>>>(W3, W3t, H2C,  YDIMC);

  // three WMMA GEMMs
  k_gemm<true,  true ><<<dim3(H1C / 128,  BATCH / 128), 128, 0, stream>>>(
      z,  W1t, b1, z1, BATCH, H1C,  XDIMC);
  k_gemm<true,  true ><<<dim3(H2C / 128,  BATCH / 128), 128, 0, stream>>>(
      z1, W2t, b2, z2, BATCH, H2C,  H1C);
  k_gemm<false, false><<<dim3(YDIMC / 128, BATCH / 128), 128, 0, stream>>>(
      z2, W3t, b3, d_out, BATCH, YDIMC, H2C);
}
